// LSTM_45878840656509
// MI455X (gfx1250) — matrix operand order repaired and checked by reading, not compiled, not run on hardware
//
#include <hip/hip_runtime.h>
#include <math.h>

// ---------------------------------------------------------------------------
// LSTM on MI455X (gfx1250): persistent bf16-WMMA recurrence.
//   gates_t = [x_t | h_{t-1}] @ [W_ih | W_hh]^T + (b_ih + b_hh)   (K = 576)
//   c_t = sig(f)*c + sig(i)*tanh(g);  h_t = sig(o)*tanh(c_t)
//   out[b,t] = b_ho + dot(h_t[b,:], W_ho)
// One persistent grid of 32 blocks loops over all T=512 steps, synchronizing
// with a device-wide atomic barrier (release fence + s_sleep spin). Cell
// state stays in LDS; only h (bf16, 128 KB) round-trips through L2 per step.
// Weight fragments are re-read from L2 every step (2.25 MB, L2-resident).
// LICM is defeated by laundering the weight *offset* (not the pointer, which
// would erase the global address space and force FLAT ops — round-3 bug).
// ---------------------------------------------------------------------------

#define B_   128
#define T_   512
#define I_   64
#define H_   512
#define KT   (I_ + H_)      // 576 concatenated K
#define AS   584            // LDS A row stride (bf16): 1168B/row, 292 words,
                            // 292 % 64 = 36 -> 16 b128 row-lanes hit 64 banks
#define NBLK 32             // persistent grid size (8 batch x 4 hidden slabs)

typedef __attribute__((ext_vector_type(16))) __bf16 v16bf;
typedef __attribute__((ext_vector_type(8)))  __bf16 v8bf;
typedef __attribute__((ext_vector_type(8)))  float  v8f;

union Frag { v16bf v; v8bf h[2]; };

__device__ __forceinline__ unsigned short f2bf_bits(float f) {
    union { float f; unsigned u; } v; v.f = f;
    unsigned r = v.u + 0x7FFFu + ((v.u >> 16) & 1u);   // RNE
    return (unsigned short)(r >> 16);
}

__device__ __forceinline__ float sigm(float x) {
    return 1.0f / (1.0f + __expf(-x));
}

// -------------------------- preprocessing ----------------------------------

__global__ __launch_bounds__(256) void prep_weights(
    const float* __restrict__ W_ih, const float* __restrict__ W_hh,
    const float* __restrict__ b_ih, const float* __restrict__ b_hh,
    unsigned short* __restrict__ Wc, float* __restrict__ bsum)
{
    int idx = blockIdx.x * 256 + threadIdx.x;
    const int total = 4 * H_ * KT;
    if (idx < total) {
        int n = idx / KT, k = idx % KT;
        float v = (k < I_) ? W_ih[(size_t)n * I_ + k]
                           : W_hh[(size_t)n * H_ + (k - I_)];
        Wc[idx] = f2bf_bits(v);
    }
    if (idx < 4 * H_) bsum[idx] = b_ih[idx] + b_hh[idx];
}

// x [B,T,I] fp32 -> xT [T,B,I] bf16
__global__ __launch_bounds__(256) void prep_x(
    const float* __restrict__ x, unsigned short* __restrict__ xT)
{
    int idx = blockIdx.x * 256 + threadIdx.x;
    if (idx >= B_ * T_ * I_) return;
    int t = idx / (B_ * I_);
    int r = idx % (B_ * I_);
    int b = r / I_, i = r % I_;
    xT[idx] = f2bf_bits(x[((size_t)b * T_ + t) * I_ + i]);
}

__global__ __launch_bounds__(256) void init_state(
    unsigned short* __restrict__ h0, unsigned* __restrict__ cnt)
{
    int idx = blockIdx.x * 256 + threadIdx.x;
    if (idx < B_ * H_) h0[idx] = 0;
    if (idx == 0) *cnt = 0u;
}

// ----------------------- device-wide step barrier --------------------------

__device__ __forceinline__ void grid_barrier(unsigned* cnt, unsigned goal) {
    __syncthreads();
    if (threadIdx.x == 0) {
        __threadfence();   // h/part stores visible at device scope
        __hip_atomic_fetch_add(cnt, 1u, __ATOMIC_RELEASE,
                               __HIP_MEMORY_SCOPE_AGENT);
        while (__hip_atomic_load(cnt, __ATOMIC_ACQUIRE,
                                 __HIP_MEMORY_SCOPE_AGENT) < goal)
            __builtin_amdgcn_s_sleep(1);
    }
    __syncthreads();
}

// --------------------------- persistent kernel -----------------------------
// grid = (8 batch-tiles of 16, 4 hidden slabs of 128), block = 256 (8 waves).
// Wave w: gate = w>>1, 64-wide column strip ((w&1)*64) -> 4 accumulators of
// 16x16, A fragment reused 4x per K-step; 72 WMMAs per wave per timestep.
__global__ __launch_bounds__(256) void lstm_persist(
    const unsigned short* __restrict__ xT,     // [T][B][64]   bf16
    const unsigned short* __restrict__ Wc,     // [2048][576]  bf16
    const float*          __restrict__ bsum,   // [2048]
    const float*          __restrict__ Who,    // [512]
    unsigned short*       __restrict__ hA,     // [B][H] bf16 (even-t input)
    unsigned short*       __restrict__ hB,     // [B][H] bf16
    float*                __restrict__ part,   // [T][B][4]
    unsigned*             __restrict__ cnt)    // barrier counter
{
    __shared__ __align__(16) unsigned short lds_a[16 * AS]; // [x_t | h] rows
    __shared__ float lds_g[4][16][128];   // gate accumulators (32 KB)
    __shared__ float lds_c[16][128];      // persistent cell state (8 KB)
    __shared__ float lds_bs[4][128];      // bias slab
    __shared__ float lds_who[128];        // W_ho slab
    __shared__ float lds_r[16][16];       // output-projection partials

    const int tid = threadIdx.x;
    const int b0  = blockIdx.x * 16;     // batch-tile origin
    const int j0  = blockIdx.y * 128;    // hidden-slab origin

    // One-time staging: zero cell state, load bias + W_ho slabs.
    for (int e = tid; e < 16 * 128; e += 256) lds_c[e >> 7][e & 127] = 0.0f;
    if (tid < 128) {
        #pragma unroll
        for (int g = 0; g < 4; ++g) lds_bs[g][tid] = bsum[g * H_ + j0 + tid];
        lds_who[tid] = Who[j0 + tid];
    }

    const int wave = tid >> 5;
    const int lane = tid & 31;
    const int half = lane >> 4;          // ISA A/B K-subrange selector
    const int l16  = lane & 15;
    const int gate = wave >> 1;
    const int n0   = gate * H_ + j0 + (wave & 1) * 64;
    const unsigned wbase = (unsigned)((n0 + l16) * KT + half * 16); // elem offset

    const int mrow  = tid >> 4;          // elementwise row (0..15)
    const int cbase = (tid & 15) * 8;    // elementwise col base (8 cols/thread)

    for (int t = 0; t < T_; ++t) {
        const unsigned short* hin  = (t & 1) ? hB : hA;
        unsigned short*       hout = (t & 1) ? hA : hB;

        // Stage A tile: 16 rows x 576 K bf16 = [x_t(64) | h_{t-1}(512)].
        for (int idx = tid; idx < 16 * (KT / 8); idx += 256) {
            int row = idx / (KT / 8);
            int k   = (idx % (KT / 8)) * 8;
            const unsigned short* src = (k < I_)
                ? xT + ((size_t)t * B_ + (b0 + row)) * I_ + k
                : hin + (size_t)(b0 + row) * H_ + (k - I_);
            *(uint4*)(&lds_a[row * AS + k]) = *(const uint4*)src;
        }
        // Prefetch next timestep's x rows (global_prefetch_b8).
        if (t + 1 < T_ && tid < 128) {
            int row = tid >> 3, k = (tid & 7) * 8;
            __builtin_prefetch(
                xT + ((size_t)(t + 1) * B_ + (b0 + row)) * I_ + k, 0, 1);
        }
        __syncthreads();

        // Launder the weight ELEMENT OFFSET (keeps Wc's global address space
        // so loads stay global_load_b128, but makes addresses loop-variant so
        // LICM cannot hoist 2.3 KB/lane of weights into scratch spills).
        unsigned woff = wbase;
        asm volatile("" : "+v"(woff));
        const unsigned short* wrt  = Wc + woff;
        const unsigned short* arow = &lds_a[l16 * AS];   // LDS AS preserved

        // GEMM: 16x64 per wave, K = 576, A reused across 4 column tiles.
        v8f acc0 = {}, acc1 = {}, acc2 = {}, acc3 = {};
        #pragma unroll
        for (int k0 = 0; k0 < KT; k0 += 32) {
            Frag a;
            a.h[0] = *(const v8bf*)(arow + k0 + half * 16);
            a.h[1] = *(const v8bf*)(arow + k0 + half * 16 + 8);
            Frag b0, b1, b2, b3;
            b0.h[0] = *(const v8bf*)(wrt + 0 * 16 * KT + k0);
            b0.h[1] = *(const v8bf*)(wrt + 0 * 16 * KT + k0 + 8);
            b1.h[0] = *(const v8bf*)(wrt + 1 * 16 * KT + k0);
            b1.h[1] = *(const v8bf*)(wrt + 1 * 16 * KT + k0 + 8);
            b2.h[0] = *(const v8bf*)(wrt + 2 * 16 * KT + k0);
            b2.h[1] = *(const v8bf*)(wrt + 2 * 16 * KT + k0 + 8);
            b3.h[0] = *(const v8bf*)(wrt + 3 * 16 * KT + k0);
            b3.h[1] = *(const v8bf*)(wrt + 3 * 16 * KT + k0 + 8);
            acc0 = __builtin_amdgcn_wmma_f32_16x16x32_bf16(
                       false, a.v, false, b0.v, (short)0, acc0, false, false);
            acc1 = __builtin_amdgcn_wmma_f32_16x16x32_bf16(
                       false, a.v, false, b1.v, (short)0, acc1, false, false);
            acc2 = __builtin_amdgcn_wmma_f32_16x16x32_bf16(
                       false, a.v, false, b2.v, (short)0, acc2, false, false);
            acc3 = __builtin_amdgcn_wmma_f32_16x16x32_bf16(
                       false, a.v, false, b3.v, (short)0, acc3, false, false);
        }

        // Spill accumulators: elem v, lane<16 -> (m=v,n=l16); lane>=16 -> m=v+8.
        {
            const int jb = (wave & 1) * 64;
            #pragma unroll
            for (int v = 0; v < 8; ++v) {
                const int m = v + 8 * half;
                lds_g[gate][m][jb +  0 + l16] = acc0[v];
                lds_g[gate][m][jb + 16 + l16] = acc1[v];
                lds_g[gate][m][jb + 32 + l16] = acc2[v];
                lds_g[gate][m][jb + 48 + l16] = acc3[v];
            }
        }
        __syncthreads();

        // Fused LSTM cell: 16x128 elements, 8 per thread; c stays in LDS.
        float hpart = 0.0f;
        #pragma unroll
        for (int u = 0; u < 8; ++u) {
            const int jl = cbase + u;
            float gi = lds_g[0][mrow][jl] + lds_bs[0][jl];
            float gf = lds_g[1][mrow][jl] + lds_bs[1][jl];
            float gg = lds_g[2][mrow][jl] + lds_bs[2][jl];
            float go = lds_g[3][mrow][jl] + lds_bs[3][jl];
            float cn = sigm(gf) * lds_c[mrow][jl] + sigm(gi) * tanhf(gg);
            float hn = sigm(go) * tanhf(cn);
            lds_c[mrow][jl] = cn;
            hout[(size_t)(b0 + mrow) * H_ + j0 + jl] = f2bf_bits(hn);
            hpart += hn * lds_who[jl];
        }
        lds_r[mrow][tid & 15] = hpart;
        __syncthreads();

        // Deterministic per-block partial of out[b,t].
        if (tid < 16) {
            float s = 0.0f;
            #pragma unroll
            for (int r = 0; r < 16; ++r) s += lds_r[tid][r];
            part[((size_t)t * B_ + (b0 + tid)) * 4 + blockIdx.y] = s;
        }

        grid_barrier(cnt, (unsigned)NBLK * (unsigned)(t + 1));
    }
}

// out[b,t] = b_ho + sum over 4 hidden-slab partials
__global__ __launch_bounds__(256) void finalize(
    const float* __restrict__ part, const float* __restrict__ b_ho,
    float* __restrict__ out)
{
    int idx = blockIdx.x * 256 + threadIdx.x;
    if (idx >= B_ * T_) return;
    int b = idx / T_, t = idx % T_;
    float s = b_ho[0];
    #pragma unroll
    for (int r = 0; r < 4; ++r)
        s += part[((size_t)t * B_ + b) * 4 + r];
    out[idx] = s;
}

// ------------------------------- launch ------------------------------------

extern "C" void kernel_launch(void* const* d_in, const int* in_sizes, int n_in,
                              void* d_out, int out_size, void* d_ws, size_t ws_size,
                              hipStream_t stream) {
    const float* x    = (const float*)d_in[0];
    const float* W_ih = (const float*)d_in[1];
    const float* W_hh = (const float*)d_in[2];
    const float* b_ih = (const float*)d_in[3];
    const float* b_hh = (const float*)d_in[4];
    const float* W_ho = (const float*)d_in[5];
    const float* b_ho = (const float*)d_in[6];
    float* out = (float*)d_out;

    char* ws = (char*)d_ws;
    size_t off = 0;
    auto carve = [&](size_t bytes) -> char* {
        char* p = ws + off;
        off = (off + bytes + 255) & ~(size_t)255;
        return p;
    };
    unsigned short* Wc   = (unsigned short*)carve((size_t)4 * H_ * KT * 2);
    float*          bsum = (float*)         carve((size_t)4 * H_ * 4);
    unsigned short* xT   = (unsigned short*)carve((size_t)T_ * B_ * I_ * 2);
    unsigned short* hA   = (unsigned short*)carve((size_t)B_ * H_ * 2);
    unsigned short* hB   = (unsigned short*)carve((size_t)B_ * H_ * 2);
    float*          part = (float*)         carve((size_t)T_ * B_ * 4 * 4);
    unsigned*       cnt  = (unsigned*)      carve(256);

    prep_weights<<<(4 * H_ * KT + 255) / 256, 256, 0, stream>>>(
        W_ih, W_hh, b_ih, b_hh, Wc, bsum);
    prep_x<<<(B_ * T_ * I_ + 255) / 256, 256, 0, stream>>>(x, xT);
    init_state<<<(B_ * H_ + 255) / 256, 256, 0, stream>>>(hA, cnt);

    lstm_persist<<<dim3(B_ / 16, H_ / 128), 256, 0, stream>>>(
        xT, Wc, bsum, W_ho, hA, hB, part, cnt);

    finalize<<<(B_ * T_ + 255) / 256, 256, 0, stream>>>(part, b_ho, out);
}
